// SSE_GDN_73976516706671
// MI455X (gfx1250) — compile-verified, matrix operand order checked by reading, not run on hardware
//
#include <hip/hip_runtime.h>
#include <hip/hip_bf16.h>
#include <stdint.h>

typedef __attribute__((ext_vector_type(16))) _Float16 v16h;
typedef __attribute__((ext_vector_type(8)))  float    v8f;

namespace {
constexpr int LTOK  = 16384;
constexpr int NSAMP = 4;
constexpr int NH    = 8;
constexpr int ND    = 128;
constexpr int NEXP  = 4;
constexpr int KSEL  = 2;
constexpr int HD    = NH * ND;      // 1024 floats per token row (q/k/v)
constexpr int L2    = LTOK * KSEL;  // 32768 destination rows

// ---- output layout (flat float32, reference return order) ----
constexpr unsigned long long OQ  = 0ull;
constexpr unsigned long long OKs = OQ  + (unsigned long long)L2 * HD;
constexpr unsigned long long OVs = OKs + (unsigned long long)L2 * HD;
constexpr unsigned long long OGK = OVs + (unsigned long long)L2 * HD;
constexpr unsigned long long OBE = OGK + (unsigned long long)L2 * NH;
constexpr unsigned long long OE  = OBE + (unsigned long long)L2 * NH;
constexpr unsigned long long OMW = OE  + (unsigned long long)LTOK * NEXP;
constexpr unsigned long long OOF = OMW + (unsigned long long)LTOK * NEXP;
constexpr unsigned long long OSS = OOF + (NSAMP * NEXP + 1);
constexpr unsigned long long OGS = OSS + (NSAMP * NEXP);

// ---- workspace layout (32-bit element offsets into d_ws) ----
constexpr int WS_E0   = 0;               // [LTOK] int   top-1 expert
constexpr int WS_E1   = WS_E0 + LTOK;    // [LTOK] int   top-2 expert
constexpr int WS_V0   = WS_E1 + LTOK;    // [LTOK] float top-1 value
constexpr int WS_V1   = WS_V0 + LTOK;    // [LTOK] float top-2 value
constexpr int WS_CNT  = WS_V1 + LTOK;    // [16]  int    pair histogram
constexpr int WS_POFF = WS_CNT + NSAMP * NEXP; // [17] int pair offsets
constexpr int WS_STOK = WS_POFF + 64;    // [L2] int   dest -> source token
constexpr int WS_SVAL = WS_STOK + L2;    // [L2] float dest -> scale
} // namespace

// ---------------- kernel 0: zero the pair histogram ----------------
__global__ void sse_init_kernel(int* wsi) {
    if (threadIdx.x < NSAMP * NEXP) wsi[WS_CNT + threadIdx.x] = 0;
}

// ------- kernel 1: softmax(N=4), top-2, e_out, mask_w, WMMA histogram -------
// Histogram trick: per wave, counts[n] = row of ones(16x32) x onehot(32x16),
// accumulated over two K-chunks with v_wmma_f32_16x16x32_f16. Reduces global
// atomics from 512/block to 32/block.
__global__ void sse_top2_kernel(const float* __restrict__ e,
                                const int* __restrict__ cu,
                                int* __restrict__ wsi, float* __restrict__ wsf,
                                float* __restrict__ out) {
    __shared__ unsigned char sel[512];           // 256 tokens * 2 selections
    const int tid = threadIdx.x;
    const int t = blockIdx.x * 256 + tid;        // grid exactly covers LTOK

    float x[NEXP];
#pragma unroll
    for (int n = 0; n < NEXP; ++n) x[n] = e[t * NEXP + n];
    float m = fmaxf(fmaxf(x[0], x[1]), fmaxf(x[2], x[3]));
    float s = 0.f;
#pragma unroll
    for (int n = 0; n < NEXP; ++n) { x[n] = expf(x[n] - m); s += x[n]; }
    float inv = 1.0f / s;
#pragma unroll
    for (int n = 0; n < NEXP; ++n) { x[n] *= inv; out[OE + (unsigned long long)t * NEXP + n] = x[n]; }

    // top-2, ties -> lower index (jax.lax.top_k semantics: strict >)
    int i0 = 0; float v0 = x[0];
#pragma unroll
    for (int n = 1; n < NEXP; ++n) if (x[n] > v0) { v0 = x[n]; i0 = n; }
    int i1 = -1; float v1 = -3.4e38f;
#pragma unroll
    for (int n = 0; n < NEXP; ++n) if (n != i0 && x[n] > v1) { v1 = x[n]; i1 = n; }

#pragma unroll
    for (int n = 0; n < NEXP; ++n)
        out[OMW + (unsigned long long)t * NEXP + n] = (n == i0 || n == i1) ? 1.0f : 0.0f;

    wsi[WS_E0 + t] = i0;  wsi[WS_E1 + t] = i1;
    wsf[WS_V0 + t] = v0;  wsf[WS_V1 + t] = v1;

    // stage this block's 512 expert selections for the WMMA histogram
    sel[tid * 2 + 0] = (unsigned char)i0;
    sel[tid * 2 + 1] = (unsigned char)i1;
    __syncthreads();

    // sample: block lies entirely inside one sample (4096 % 256 == 0)
    int t0 = blockIdx.x * 256;
    int smp = 0;
#pragma unroll
    for (int i = 1; i <= NSAMP; ++i) if (cu[i] <= t0) smp = i;

    // ---- WMMA histogram: D = ones(16x32) x onehot(32x16) + D, 2 chunks/wave ----
    const int lane = tid & 31;
    const int wave = tid >> 5;
    const int col  = lane & 15;      // B-matrix column N (expert id if < 4)
    const int half = lane >> 4;      // K-half: 0 -> K 0..15, 1 -> K 16..31

    v16h aones;
#pragma unroll
    for (int i = 0; i < 16; ++i) aones[i] = (_Float16)1.0f;

    v8f acc = {};
#pragma unroll
    for (int c = 0; c < 2; ++c) {                // two 32-selection chunks
        const int kbase = wave * 64 + c * 32 + half * 16;
        v16h b;
#pragma unroll
        for (int j = 0; j < 8; ++j) {            // VGPR j packs K=2j, 2j+1
            b[2 * j + 0] = (sel[kbase + 2 * j + 0] == col) ? (_Float16)1.0f : (_Float16)0.0f;
            b[2 * j + 1] = (sel[kbase + 2 * j + 1] == col) ? (_Float16)1.0f : (_Float16)0.0f;
        }
        acc = __builtin_amdgcn_wmma_f32_16x16x32_f16(
            false, aones, false, b, (short)0, acc, false, false);
    }
    // D layout: VGPR0, lanes 0..15 -> row M=0, N=lane. Lane n<4 holds count[n].
    if (lane < NEXP) {
        int cnt = (int)(acc[0] + 0.5f);
        atomicAdd(&wsi[WS_CNT + smp * NEXP + lane], cnt);
    }
}

// ------- kernel 2: 16-wide exclusive scan -> offsets / state_sizes -------
__global__ void sse_scan_kernel(int* __restrict__ wsi, float* __restrict__ out) {
    if (threadIdx.x != 0) return;
    int acc = 0;
    wsi[WS_POFF + 0] = 0;
    out[OOF + 0] = 0.0f;
    for (int i = 0; i < NSAMP * NEXP; ++i) {
        int c = wsi[WS_CNT + i];
        out[OSS + i] = (float)c;
        acc += c;
        wsi[WS_POFF + i + 1] = acc;
        out[OOF + i + 1] = (float)acc;
    }
}

// ------- kernel 3: stable rank assignment (counting sort placement) -------
// one block per sample; wave32 ballot scan + cross-wave LDS scan
__global__ void sse_rank_kernel(const int* __restrict__ cu,
                                int* __restrict__ wsi, float* __restrict__ wsf,
                                float* __restrict__ out) {
    __shared__ int totals[NEXP][8];
    __shared__ int base[NEXP];
    const int smp  = blockIdx.x;
    const int lo   = cu[smp], hi = cu[smp + 1];
    const int tid  = threadIdx.x;
    const int lane = tid & 31;
    const int wave = tid >> 5;

    if (tid < NEXP) base[tid] = wsi[WS_POFF + smp * NEXP + tid];
    __syncthreads();

    for (int c = lo; c < hi; c += 256) {
        int t = c + tid;
        bool valid = (t < hi);
        int e0 = valid ? wsi[WS_E0 + t] : -1;
        int e1 = valid ? wsi[WS_E1 + t] : -1;

        int rank[NEXP];
#pragma unroll
        for (int n = 0; n < NEXP; ++n) {
            unsigned b = (unsigned)__ballot(valid && (e0 == n || e1 == n));
            unsigned lt = (lane == 0) ? 0u : ((~0u) >> (32 - lane));
            rank[n] = __popc(b & lt);
            if (lane == 0) totals[n][wave] = __popc(b);
        }
        __syncthreads();
#pragma unroll
        for (int n = 0; n < NEXP; ++n) {
            int pre = 0;
            for (int w = 0; w < wave; ++w) pre += totals[n][w];
            rank[n] += base[n] + pre;
        }
        if (valid) {
            int d0 = rank[e0];
            wsi[WS_STOK + d0] = t;
            wsf[WS_SVAL + d0] = wsf[WS_V0 + t];
            out[OGS + d0] = (float)t;
            int d1 = rank[e1];
            wsi[WS_STOK + d1] = t;
            wsf[WS_SVAL + d1] = wsf[WS_V1 + t];
            out[OGS + d1] = (float)t;
        }
        __syncthreads();
        if (tid < NEXP) {
            int all = 0;
            for (int w = 0; w < 8; ++w) all += totals[tid][w];
            base[tid] += all;
        }
        __syncthreads();
    }
}

// ------- kernel 4: bandwidth gather. One block (256 thr) per dest row -------
// q/k rows: b128 load -> scale -> b128 store.  v row: CDNA5 async path
// global_load_async_to_lds_b128 + s_wait_asynccnt + global_store_async_from_lds_b128
__global__ void sse_gather_kernel(const float* __restrict__ q,
                                  const float* __restrict__ k,
                                  const float* __restrict__ v,
                                  const float* __restrict__ gk,
                                  const float* __restrict__ beta,
                                  const int* __restrict__ wsi,
                                  const float* __restrict__ wsf,
                                  float* __restrict__ out) {
    __shared__ __align__(16) char vbuf[256 * 16];   // 4 KB = one v row
    const int j   = blockIdx.x;         // destination row [0, 2L)
    const int tid = threadIdx.x;

    const int   t  = wsi[WS_STOK + j];
    const float sc = wsf[WS_SVAL + j];

    const unsigned long long srcRow = (unsigned long long)t * HD;
    const unsigned long long dstRow = (unsigned long long)j * HD;

    // ---- q (scaled) ----
    {
        float4 r = reinterpret_cast<const float4*>(q + srcRow)[tid];
        r.x *= sc; r.y *= sc; r.z *= sc; r.w *= sc;
        reinterpret_cast<float4*>(out + OQ + dstRow)[tid] = r;
    }
    // ---- k (scaled) ----
    {
        float4 r = reinterpret_cast<const float4*>(k + srcRow)[tid];
        r.x *= sc; r.y *= sc; r.z *= sc; r.w *= sc;
        reinterpret_cast<float4*>(out + OKs + dstRow)[tid] = r;
    }
    // ---- v (unscaled): async global -> LDS -> global, no VGPR round-trip ----
    {
        unsigned lds = (unsigned)(unsigned long long)(uintptr_t)(&vbuf[tid * 16]);
        unsigned long long gsrc = (unsigned long long)(uintptr_t)(v + srcRow) + (unsigned long long)tid * 16;
        asm volatile("global_load_async_to_lds_b128 %0, %1, off"
                     :: "v"(lds), "v"(gsrc) : "memory");
        __builtin_amdgcn_s_wait_asynccnt(0);   // lane's own load -> LDS complete
        unsigned long long gdst = (unsigned long long)(uintptr_t)(out + OVs + dstRow) + (unsigned long long)tid * 16;
        asm volatile("global_store_async_from_lds_b128 %0, %1, off"
                     :: "v"(gdst), "v"(lds) : "memory");
        // s_endpgm implicitly waits all counters; LDS is private to this block's row
    }
    // ---- gk / beta (8 floats each) ----
    if (tid < NH) {
        out[OGK + (unsigned long long)j * NH + tid] = gk[(unsigned long long)t * NH + tid];
    } else if (tid < 2 * NH) {
        int h = tid - NH;
        out[OBE + (unsigned long long)j * NH + h] = beta[(unsigned long long)t * NH + h];
    }
}

extern "C" void kernel_launch(void* const* d_in, const int* in_sizes, int n_in,
                              void* d_out, int out_size, void* d_ws, size_t ws_size,
                              hipStream_t stream) {
    const float* q    = (const float*)d_in[0];
    const float* k    = (const float*)d_in[1];
    const float* v    = (const float*)d_in[2];
    const float* gk   = (const float*)d_in[3];
    const float* beta = (const float*)d_in[4];
    const float* e    = (const float*)d_in[5];
    const int*   cu   = (const int*)d_in[6];
    // d_in[7] = K (==2), compile-time constant here

    float* out = (float*)d_out;
    int*   wsi = (int*)d_ws;
    float* wsf = (float*)d_ws;

    sse_init_kernel<<<1, 32, 0, stream>>>(wsi);
    sse_top2_kernel<<<LTOK / 256, 256, 0, stream>>>(e, cu, wsi, wsf, out);
    sse_scan_kernel<<<1, 32, 0, stream>>>(wsi, out);
    sse_rank_kernel<<<NSAMP, 256, 0, stream>>>(cu, wsi, wsf, out);
    sse_gather_kernel<<<L2, 256, 0, stream>>>(q, k, v, gk, beta, wsi, wsf, out);
}
